// DAE_Model_54322746360315
// MI455X (gfx1250) — compile-verified, matrix operand order checked by reading, not run on hardware
//
#include <hip/hip_runtime.h>

typedef __attribute__((ext_vector_type(16))) _Float16 v16h;
typedef __attribute__((ext_vector_type(8)))  _Float16 v8h;
typedef __attribute__((ext_vector_type(8)))  float    v8f;

namespace {
constexpr int kB = 512, kT = 400;
constexpr int kX = 12, kZ = 6, kV = 4, kI = 4, kH = 128, kAll = 26;  // 12+6+4+4
constexpr int kDeIn = 78, kDeKP = 96;   // 3*26 padded to mult of 32
constexpr int kAeIn = 48, kAeKP = 64;   // 26+12+6+4 padded
constexpr int kInIn = 14, kInKP = 32;   // 6+4+4 padded
constexpr int kRows = 16;               // batch rows per wave; 1 wave per block
}

// ---- fragment load from LDS ----------------------------------------------
// A (16xK f16): lane L holds row m=L%16; halves e=0..7 -> K = k0 + (L/16)*8 + e
//                                        halves e=8..15 -> K = k0 + 16 + (L/16)*8 + (e-8)
// B is stored as W^T[n][k], so the same gather with row=n gives the B fragment.
template<int KP>
__device__ __forceinline__ v16h frag_ld(const _Float16* __restrict__ base, int row,
                                        int k0, int lane) {
  const int kh = lane >> 4;
  const _Float16* p = base + row * KP + k0 + kh * 8;
  v8h lo = *(const v8h*)p;          // ds_load_b128
  v8h hi = *(const v8h*)(p + 16);   // ds_load_b128
  v16h f;
#pragma unroll
  for (int e = 0; e < 8; ++e) { f[e] = lo[e]; f[8 + e] = hi[e]; }
  return f;
}

// hidden layer: [16 x KP] @ [KP x 128] + bias, ELU, write f16 to out[16][128]
// A fragments hoisted: loaded once per layer, reused for all 8 N-tiles.
template<int KP>
__device__ __forceinline__ void layer_hidden(const _Float16* __restrict__ in,
                                             const _Float16* __restrict__ wt,
                                             const float* __restrict__ bias,
                                             _Float16* __restrict__ out, int lane) {
  constexpr int NK = KP / 32;
  const int n  = lane & 15;
  const int kh = lane >> 4;
  v16h a[NK];
#pragma unroll
  for (int kt = 0; kt < NK; ++kt) a[kt] = frag_ld<KP>(in, lane & 15, kt * 32, lane);
#pragma unroll
  for (int nt = 0; nt < 8; ++nt) {
    const float bv = bias[nt * 16 + n];
    v8f acc = {bv, bv, bv, bv, bv, bv, bv, bv};   // bias as the C matrix
#pragma unroll
    for (int kt = 0; kt < NK; ++kt) {
      v16h b = frag_ld<KP>(wt + nt * 16 * KP, n, kt * 32, lane);
      acc = __builtin_amdgcn_wmma_f32_16x16x32_f16(false, a[kt], false, b,
                                                   (short)0, acc, false, false);
    }
#pragma unroll
    for (int r = 0; r < 8; ++r) {            // D layout: m = 8*(L/16)+r, n = L%16
      float v = acc[r];
      v = v > 0.f ? v : (__expf(v) - 1.f);   // ELU
      out[(kh * 8 + r) * kH + nt * 16 + n] = (_Float16)v;
    }
  }
}

// output layer: [16 x KP] @ [KP x 16] + bias (linear), write f32 to out[16][16]
template<int KP>
__device__ __forceinline__ void layer_out(const _Float16* __restrict__ in,
                                          const _Float16* __restrict__ wt,
                                          const float* __restrict__ bias,
                                          float* __restrict__ out, int lane) {
  const int n  = lane & 15;
  const int kh = lane >> 4;
  const float bv = bias[n];
  v8f acc = {bv, bv, bv, bv, bv, bv, bv, bv};
#pragma unroll
  for (int kt = 0; kt < KP; kt += 32) {
    v16h a = frag_ld<KP>(in, lane & 15, kt, lane);
    v16h b = frag_ld<KP>(wt, n, kt, lane);
    acc = __builtin_amdgcn_wmma_f32_16x16x32_f16(false, a, false, b,
                                                 (short)0, acc, false, false);
  }
#pragma unroll
  for (int r = 0; r < 8; ++r) out[(kh * 8 + r) * 16 + n] = acc[r];
}

// cooperative weight load: global f32 [Kin][Nout] -> LDS f16 W^T [Npad][Kpad], zero-padded
__device__ void load_wT(const float* __restrict__ W, int Kin, int Nout, int Kpad,
                        int Npad, _Float16* __restrict__ dst, int tid) {
  const int tot = Npad * Kpad;
  for (int idx = tid; idx < tot; idx += 32) {
    const int n = idx / Kpad, k = idx % Kpad;
    dst[idx] = (_Float16)((n < Nout && k < Kin) ? W[k * Nout + n] : 0.0f);
  }
}
__device__ void load_b(const float* __restrict__ b, int Nout, int Npad,
                       float* __restrict__ dst, int tid) {
  for (int n = tid; n < Npad; n += 32) dst[n] = (n < Nout) ? b[n] : 0.0f;
}

__global__ __launch_bounds__(32)
void dae_scan_kernel(const float* __restrict__ tG, const float* __restrict__ zG,
                     const float* __restrict__ vG, const float* __restrict__ iG,
                     const float* __restrict__ evT, const float* __restrict__ zJ,
                     const float* __restrict__ vJ,
                     const float* iW0, const float* iB0, const float* iW1,
                     const float* iB1, const float* iW2, const float* iB2,
                     const float* dW0, const float* dB0, const float* dW1,
                     const float* dB1, const float* dW2, const float* dB2,
                     const float* dW3, const float* dB3,
                     const float* aW0, const float* aB0, const float* aW1,
                     const float* aB1, const float* aW2, const float* aB2,
                     const float* aW3, const float* aB3,
                     float* __restrict__ outX, float* __restrict__ outI) {
  // ---- LDS: resident weights (f16, transposed, padded) --------------------
  __shared__ __align__(16) _Float16 sDe0[kH * kDeKP];   // also holds init W0 in phase 0
  __shared__ __align__(16) _Float16 sDe1[kH * kH];      // also init W1
  __shared__ __align__(16) _Float16 sDe2[kH * kH];
  __shared__ __align__(16) _Float16 sDe3[16 * kH];      // also init W2 (Npad=16)
  __shared__ float sBde[3 * kH + 16];
  __shared__ __align__(16) _Float16 sAe0[kH * kAeKP];
  __shared__ __align__(16) _Float16 sAe1[kH * kH];
  __shared__ __align__(16) _Float16 sAe2[kH * kH];
  __shared__ __align__(16) _Float16 sAe3[16 * kH];
  __shared__ float sBae[3 * kH + 16];
  // ---- per-wave working buffers (single wave per block) -------------------
  __shared__ __align__(16) _Float16 sA0[kRows * kH];
  __shared__ __align__(16) _Float16 sA1[kRows * kH];
  __shared__ float sAll[kRows * kAll];
  __shared__ float sXs[kRows * kX];
  __shared__ float sIs[kRows * kI];
  __shared__ float sO[kRows * 16];

  const int tid  = threadIdx.x;
  const int lane = tid & 31;
  const int rowB = blockIdx.x * kRows;

  // ================= phase 0: ae + init weights ============================
  load_wT(aW0, kAeIn, kH, kAeKP, kH, sAe0, tid);
  load_wT(aW1, kH, kH, kH, kH, sAe1, tid);
  load_wT(aW2, kH, kH, kH, kH, sAe2, tid);
  load_wT(aW3, kH, kI, kH, 16, sAe3, tid);
  load_b(aB0, kH, kH, sBae, tid);
  load_b(aB1, kH, kH, sBae + kH, tid);
  load_b(aB2, kH, kH, sBae + 2 * kH, tid);
  load_b(aB3, kI, 16, sBae + 3 * kH, tid);
  load_wT(iW0, kInIn, kH, kInKP, kH, sDe0, tid);   // init overlays de region
  load_wT(iW1, kH, kH, kH, kH, sDe1, tid);
  load_wT(iW2, kH, kX, kH, 16, sDe3, tid);
  load_b(iB0, kH, kH, sBde, tid);
  load_b(iB1, kH, kH, sBde + kH, tid);
  load_b(iB2, kX, 16, sBde + 3 * kH, tid);
  __syncthreads();

  // x0 = init_mlp([z0, v0, i0])
  for (int idx = lane; idx < kRows * kInKP; idx += 32) {
    const int m = idx / kInKP, c = idx % kInKP;
    const size_t b = (size_t)(rowB + m);
    float v = 0.f;
    if (c < kZ)               v = zG[b * kT * kZ + c];
    else if (c < kZ + kV)     v = vG[b * kT * kV + (c - kZ)];
    else if (c < kInIn)       v = iG[b * kT * kI + (c - kZ - kV)];
    sA0[idx] = (_Float16)v;
  }
  __syncthreads();
  layer_hidden<kInKP>(sA0, sDe0, sBde, sA1, lane);          __syncthreads();
  layer_hidden<kH>(sA1, sDe1, sBde + kH, sA0, lane);        __syncthreads();
  layer_out<kH>(sA0, sDe3, sBde + 3 * kH, sO, lane);        __syncthreads();

  // all_init = [x0, z0, v0, i0]; emit x0
  for (int idx = lane; idx < kRows * kAll; idx += 32) {
    const int m = idx / kAll, c = idx % kAll;
    const size_t b = (size_t)(rowB + m);
    float v;
    if (c < kX)                    v = sO[m * 16 + c];
    else if (c < kX + kZ)          v = zG[b * kT * kZ + (c - kX)];
    else if (c < kX + kZ + kV)     v = vG[b * kT * kV + (c - kX - kZ)];
    else                           v = iG[b * kT * kI + (c - kX - kZ - kV)];
    sAll[idx] = v;
    if (c < kX) { sXs[m * kX + c] = v; outX[(b * kT) * kX + c] = v; }
  }
  __syncthreads();

  // i0_hat = ae_mlp([all_init, x0, z0, v0])
  for (int idx = lane; idx < kRows * kAeKP; idx += 32) {
    const int m = idx / kAeKP, c = idx % kAeKP;
    const size_t b = (size_t)(rowB + m);
    float v = 0.f;
    if (c < kAll)                  v = sAll[m * kAll + c];
    else if (c < kAll + kX)        v = sXs[m * kX + (c - kAll)];
    else if (c < kAll + kX + kZ)   v = zG[b * kT * kZ + (c - kAll - kX)];
    else if (c < kAeIn)            v = vG[b * kT * kV + (c - kAll - kX - kZ)];
    sA0[idx] = (_Float16)v;
  }
  __syncthreads();
  layer_hidden<kAeKP>(sA0, sAe0, sBae, sA1, lane);          __syncthreads();
  layer_hidden<kH>(sA1, sAe1, sBae + kH, sA0, lane);        __syncthreads();
  layer_hidden<kH>(sA0, sAe2, sBae + 2 * kH, sA1, lane);    __syncthreads();
  layer_out<kH>(sA1, sAe3, sBae + 3 * kH, sO, lane);        __syncthreads();
  for (int idx = lane; idx < kRows * kI; idx += 32) {
    const int m = idx / kI, c = idx % kI;
    const size_t b = (size_t)(rowB + m);
    const float v = sO[m * 16 + c];
    sIs[idx] = v;
    outI[(b * kT) * kI + c] = v;
  }
  __syncthreads();

  // ================= phase 1: de weights (overwrite init) ==================
  load_wT(dW0, kDeIn, kH, kDeKP, kH, sDe0, tid);
  load_wT(dW1, kH, kH, kH, kH, sDe1, tid);
  load_wT(dW2, kH, kH, kH, kH, sDe2, tid);
  load_wT(dW3, kH, kX, kH, 16, sDe3, tid);
  load_b(dB0, kH, kH, sBde, tid);
  load_b(dB1, kH, kH, sBde + kH, tid);
  load_b(dB2, kH, kH, sBde + 2 * kH, tid);
  load_b(dB3, kX, 16, sBde + 3 * kH, tid);
  __syncthreads();

  // ================= sequential scan over time =============================
  for (int k = 0; k < kT - 1; ++k) {
    // de input = [all_init, xt_all - all_init, xt_all], xt_all = [x,zt,vt,i]
    for (int idx = lane; idx < kRows * kDeKP; idx += 32) {
      const int m = idx / kDeKP, c = idx % kDeKP;
      const size_t b = (size_t)(rowB + m);
      float v = 0.f;
      if (c < kDeIn) {
        const int seg = c / kAll, cc = c % kAll;
        const float ai = sAll[m * kAll + cc];
        if (seg == 0) v = ai;
        else {
          float xa;
          if (cc < kX)               xa = sXs[m * kX + cc];
          else if (cc < kX + kZ) {
            const float tk  = tG[b * kT + k];
            const float tk1 = tG[b * kT + k + 1];
            const float ev  = evT[b];
            const bool  msk = (ev > tk) && (ev <= tk1);
            const float zk  = zG[(b * kT + k) * kZ + (cc - kX)];
            xa = msk ? zJ[b * kZ + (cc - kX)] : zk;
          } else if (cc < kX + kZ + kV) {
            const float tk  = tG[b * kT + k];
            const float tk1 = tG[b * kT + k + 1];
            const float ev  = evT[b];
            const bool  msk = (ev > tk) && (ev <= tk1);
            const float vk  = vG[(b * kT + k) * kV + (cc - kX - kZ)];
            xa = msk ? vJ[b * kV + (cc - kX - kZ)] : vk;
          } else                     xa = sIs[m * kI + (cc - kX - kZ - kV)];
          v = (seg == 1) ? (xa - ai) : xa;
        }
      }
      sA0[idx] = (_Float16)v;
    }
    __syncthreads();
    layer_hidden<kDeKP>(sA0, sDe0, sBde, sA1, lane);        __syncthreads();
    layer_hidden<kH>(sA1, sDe1, sBde + kH, sA0, lane);      __syncthreads();
    layer_hidden<kH>(sA0, sDe2, sBde + 2 * kH, sA1, lane);  __syncthreads();
    layer_out<kH>(sA1, sDe3, sBde + 3 * kH, sO, lane);      __syncthreads();

    // x_{k+1} = x_k + dt * dx ; emit
    for (int idx = lane; idx < kRows * kX; idx += 32) {
      const int m = idx / kX, c = idx % kX;
      const size_t b = (size_t)(rowB + m);
      const float dt = tG[b * kT + k + 1] - tG[b * kT + k];
      const float xn = sXs[idx] + dt * sO[m * 16 + c];
      sXs[idx] = xn;
      outX[(b * kT + k + 1) * kX + c] = xn;
    }
    __syncthreads();

    // ae input = [all_init, x_{k+1}, z_{k+1}, v_{k+1}]
    for (int idx = lane; idx < kRows * kAeKP; idx += 32) {
      const int m = idx / kAeKP, c = idx % kAeKP;
      const size_t b = (size_t)(rowB + m);
      float v = 0.f;
      if (c < kAll)                  v = sAll[m * kAll + c];
      else if (c < kAll + kX)        v = sXs[m * kX + (c - kAll)];
      else if (c < kAll + kX + kZ)   v = zG[(b * kT + k + 1) * kZ + (c - kAll - kX)];
      else if (c < kAeIn)            v = vG[(b * kT + k + 1) * kV + (c - kAll - kX - kZ)];
      sA0[idx] = (_Float16)v;
    }
    __syncthreads();
    layer_hidden<kAeKP>(sA0, sAe0, sBae, sA1, lane);        __syncthreads();
    layer_hidden<kH>(sA1, sAe1, sBae + kH, sA0, lane);      __syncthreads();
    layer_hidden<kH>(sA0, sAe2, sBae + 2 * kH, sA1, lane);  __syncthreads();
    layer_out<kH>(sA1, sAe3, sBae + 3 * kH, sO, lane);      __syncthreads();

    for (int idx = lane; idx < kRows * kI; idx += 32) {
      const int m = idx / kI, c = idx % kI;
      const size_t b = (size_t)(rowB + m);
      const float v = sO[m * 16 + c];
      sIs[idx] = v;
      outI[(b * kT + k + 1) * kI + c] = v;
    }
    __syncthreads();
  }
}

extern "C" void kernel_launch(void* const* d_in, const int* in_sizes, int n_in,
                              void* d_out, int out_size, void* d_ws, size_t ws_size,
                              hipStream_t stream) {
  (void)in_sizes; (void)n_in; (void)out_size; (void)d_ws; (void)ws_size;
  const float* tG  = (const float*)d_in[0];
  // d_in[1] is x: unused by the reference computation
  const float* zG  = (const float*)d_in[2];
  const float* vG  = (const float*)d_in[3];
  const float* iG  = (const float*)d_in[4];
  const float* evT = (const float*)d_in[5];
  const float* zJ  = (const float*)d_in[6];
  const float* vJ  = (const float*)d_in[7];
  const float* iW0 = (const float*)d_in[8];
  const float* iB0 = (const float*)d_in[9];
  const float* iW1 = (const float*)d_in[10];
  const float* iB1 = (const float*)d_in[11];
  const float* iW2 = (const float*)d_in[12];
  const float* iB2 = (const float*)d_in[13];
  const float* dW0 = (const float*)d_in[14];
  const float* dB0 = (const float*)d_in[15];
  const float* dW1 = (const float*)d_in[16];
  const float* dB1 = (const float*)d_in[17];
  const float* dW2 = (const float*)d_in[18];
  const float* dB2 = (const float*)d_in[19];
  const float* dW3 = (const float*)d_in[20];
  const float* dB3 = (const float*)d_in[21];
  const float* aW0 = (const float*)d_in[22];
  const float* aB0 = (const float*)d_in[23];
  const float* aW1 = (const float*)d_in[24];
  const float* aB1 = (const float*)d_in[25];
  const float* aW2 = (const float*)d_in[26];
  const float* aB2 = (const float*)d_in[27];
  const float* aW3 = (const float*)d_in[28];
  const float* aB3 = (const float*)d_in[29];

  float* outX = (float*)d_out;                       // (B, T, 12)
  float* outI = outX + (size_t)kB * kT * kX;         // (B, T, 4)

  dim3 grid(kB / kRows), block(32);                  // 1 wave per block, 32 WGPs
  dae_scan_kernel<<<grid, block, 0, stream>>>(
      tG, zG, vG, iG, evT, zJ, vJ,
      iW0, iB0, iW1, iB1, iW2, iB2,
      dW0, dB0, dW1, dB1, dW2, dB2, dW3, dB3,
      aW0, aB0, aW1, aB1, aW2, aB2, aW3, aB3,
      outX, outI);
}
